// GINEncoder_25460566130972
// MI455X (gfx1250) — compile-verified, matrix operand order checked by reading, not run on hardware
//
#include <hip/hip_runtime.h>

// GIN encoder: N=50000 nodes, D=512, L=3 layers, E=800000 edges.
// MI455X / gfx1250 (wave32) strategy:
//  - GEMMs via V_WMMA_F32_16X16X32_BF16. W is pre-packed to bf16 in fragment
//    order once per layer (L2-resident 512KB) -> B frag = 2x global_load_b128.
//    A slab (16x512, with fused h+agg) staged to LDS as bf16 -> A frag =
//    2x ds_load_b128, bank-conflict-free. Zero cvts in the K-loop.
//  - Edge aggregation via global f32 atomics (~3.2GB + 410M atomics per layer).
//  - BatchNorm: two-pass (atomic column sums/sumsq, then fused normalize+ReLU).
//  - d_out doubles as the h ping buffer; d_ws holds Wp + agg + hidden + stats.

#define NN 50000
#define DDIM 512
#define NL 3
#define EE 800000
#define NDTOT (NN * DDIM)
#define LDS_BF_STRIDE 520   // bf16 units; 520*2/4 = 260 dwords, 260%64==4 -> clean banks

typedef __attribute__((ext_vector_type(16))) __bf16 v16bf;
typedef __attribute__((ext_vector_type(8)))  __bf16 v8bf;
typedef __attribute__((ext_vector_type(4)))  __bf16 v4bf;
typedef __attribute__((ext_vector_type(8)))  float  v8f;

// ---------------------------------------------------------------- zero fill
__global__ void k_zero(float* __restrict__ p, int n) {
    int i = blockIdx.x * blockDim.x + threadIdx.x;
    int stride = gridDim.x * blockDim.x;
    for (; i < n; i += stride) p[i] = 0.0f;
}

// ------------------------------------------------- edge scatter-add (GIN agg)
__global__ void k_scatter(const float* __restrict__ h,
                          const int* __restrict__ srcI,
                          const int* __restrict__ dstI,
                          float* __restrict__ agg) {
    const long long total = (long long)EE * (DDIM / 4);  // 128 float4 per edge
    long long i = (long long)blockIdx.x * blockDim.x + threadIdx.x;
    const long long stride = (long long)gridDim.x * blockDim.x;
    for (; i < total; i += stride) {
        const int e = (int)(i >> 7);
        const int c = ((int)i & 127) << 2;
        const float4 v = *(const float4*)(h + (long long)srcI[e] * DDIM + c);
        float* a = agg + (long long)dstI[e] * DDIM + c;
        atomicAdd(a + 0, v.x);
        atomicAdd(a + 1, v.y);
        atomicAdd(a + 2, v.z);
        atomicAdd(a + 3, v.w);
    }
}

// ---------------------------------------------- pack W (fp32) -> bf16 B-frags
// Group g = (chunk*512 + n)*2 + lh holds the 16 bf16 a lane (half lh, col n)
// feeds to WMMA for K-chunk `chunk`: elements 0..7 -> K=k0+lh*8+e,
// elements 8..15 -> K=k0+16+lh*8+(e-8).
__global__ void k_packW(const float* __restrict__ W, __bf16* __restrict__ Wp) {
    const int g = blockIdx.x * blockDim.x + threadIdx.x;
    if (g >= (DDIM / 32) * DDIM * 2) return;
    const int lh = g & 1;
    const int n  = (g >> 1) & (DDIM - 1);
    const int ch = g >> 10;
    const int k0 = ch * 32;
    __bf16* o = Wp + (size_t)g * 16;
#pragma unroll
    for (int e = 0; e < 16; ++e) {
        const int kk = (e < 8) ? (lh * 8 + e) : (16 + lh * 8 + (e - 8));
        o[e] = (__bf16)W[(size_t)(k0 + kk) * DDIM + n];
    }
}

// ------------------------------------------------------------- WMMA GEMM
// out[M,512] = (A (+ A2)) @ W + bias, W pre-packed bf16.  Block = 8 waves
// handles a 16x512 slab; wave w covers cols [w*64, w*64+64) (4 tiles).
template <int HASA2>
__global__ __launch_bounds__(256)
void k_gemm_bf16(const float* __restrict__ A,
                 const float* __restrict__ A2,       // GIN aggregation addend
                 const __bf16* __restrict__ Wp,      // packed [16][512][2][16]
                 const float* __restrict__ bias,     // [512]
                 float* __restrict__ out) {
    __shared__ __bf16 As[16 * LDS_BF_STRIDE];        // ~16.6 KB

    const int row0 = blockIdx.x * 16;

    // Phase 1: coalesced stage of A (+A2) -> bf16 LDS (row-major).
    for (int idx = threadIdx.x; idx < 16 * (DDIM / 4); idx += 256) {
        const int r  = idx >> 7;                     // 128 float4 per row
        const int c4 = (idx & 127) << 2;
        float4 v = *(const float4*)(A + (long long)(row0 + r) * DDIM + c4);
        if (HASA2) {
            const float4 g = *(const float4*)(A2 + (long long)(row0 + r) * DDIM + c4);
            v.x += g.x; v.y += g.y; v.z += g.z; v.w += g.w;
        }
        v4bf o;
        o[0] = (__bf16)v.x; o[1] = (__bf16)v.y; o[2] = (__bf16)v.z; o[3] = (__bf16)v.w;
        *(v4bf*)(&As[r * LDS_BF_STRIDE + c4]) = o;   // ds_store_b64
    }
    __syncthreads();

    // Phase 2: tensor loop. 2 ds_load_b128 + 8 global_load_b128 + 4 WMMA / step.
    const int lane = threadIdx.x & 31;
    const int wave = threadIdx.x >> 5;
    const int lm = lane & 15;                        // A row / B,C column within tile
    const int lh = lane >> 4;                        // lane half
    const int colbase = wave * 64 + lm;

    v8f c0, c1, c2, c3;
    {
        const float b0 = bias[colbase];
        const float b1 = bias[colbase + 16];
        const float b2 = bias[colbase + 32];
        const float b3 = bias[colbase + 48];
#pragma unroll
        for (int v = 0; v < 8; ++v) { c0[v] = b0; c1[v] = b1; c2[v] = b2; c3[v] = b3; }
    }

    const __bf16* AsRow = &As[lm * LDS_BF_STRIDE];

    for (int ch = 0; ch < DDIM / 32; ++ch) {
        const int k0 = ch * 32;
        if (ch + 1 < DDIM / 32)                      // global_prefetch_b8 next Wp slab
            __builtin_prefetch(Wp + ((size_t)((ch + 1) * DDIM + colbase) * 2 + lh) * 16, 0, 1);

        // A fragment: two contiguous 8-element bf16 runs (conflict-free b128s).
        const v8bf alo = *(const v8bf*)(AsRow + k0 + lh * 8);
        const v8bf ahi = *(const v8bf*)(AsRow + k0 + 16 + lh * 8);
        const v16bf a  = __builtin_shufflevector(alo, ahi,
                            0, 1, 2, 3, 4, 5, 6, 7, 8, 9, 10, 11, 12, 13, 14, 15);

        // B fragments: pre-packed, contiguous 32B per tile column group.
        const __bf16* wp = Wp + ((size_t)(ch * DDIM + colbase) * 2 + lh) * 16;
        const v16bf b0 = *(const v16bf*)(wp);
        const v16bf b1 = *(const v16bf*)(wp + 512);    // +16 columns
        const v16bf b2 = *(const v16bf*)(wp + 1024);   // +32 columns
        const v16bf b3 = *(const v16bf*)(wp + 1536);   // +48 columns

        c0 = __builtin_amdgcn_wmma_f32_16x16x32_bf16(false, a, false, b0, (short)0, c0, false, false);
        c1 = __builtin_amdgcn_wmma_f32_16x16x32_bf16(false, a, false, b1, (short)0, c1, false, false);
        c2 = __builtin_amdgcn_wmma_f32_16x16x32_bf16(false, a, false, b2, (short)0, c2, false, false);
        c3 = __builtin_amdgcn_wmma_f32_16x16x32_bf16(false, a, false, b3, (short)0, c3, false, false);
    }

    // C/D store: VGPR v -> row row0 + lh*8 + v, columns colbase + {0,16,32,48}.
    float* o0 = out + (long long)(row0 + lh * 8) * DDIM + colbase;
#pragma unroll
    for (int v = 0; v < 8; ++v) {
        o0[(long long)v * DDIM]      = c0[v];
        o0[(long long)v * DDIM + 16] = c1[v];
        o0[(long long)v * DDIM + 32] = c2[v];
        o0[(long long)v * DDIM + 48] = c3[v];
    }
}

// ------------------------------------------- per-column sum / sum-of-squares
__global__ __launch_bounds__(256)
void k_colstats(const float* __restrict__ x,
                float* __restrict__ csum, float* __restrict__ csq) {
    const int c0 = threadIdx.x * 2;
    float s0 = 0.f, s1 = 0.f, q0 = 0.f, q1 = 0.f;
    for (int r = blockIdx.x; r < NN; r += gridDim.x) {
        const float2 v = *(const float2*)(x + (long long)r * DDIM + c0);
        s0 += v.x; q0 += v.x * v.x;
        s1 += v.y; q1 += v.y * v.y;
    }
    atomicAdd(&csum[c0],     s0);
    atomicAdd(&csum[c0 + 1], s1);
    atomicAdd(&csq[c0],      q0);
    atomicAdd(&csq[c0 + 1],  q1);
}

// --------------------------------------------- batchnorm apply (+ optional ReLU)
__global__ void k_bn(const float* __restrict__ src, float* __restrict__ dst,
                     const float* __restrict__ csum, const float* __restrict__ csq,
                     const float* __restrict__ g, const float* __restrict__ b,
                     int relu) {
    const float inv_n = 1.0f / (float)NN;
    long long i = (long long)blockIdx.x * blockDim.x + threadIdx.x;
    const long long stride = (long long)gridDim.x * blockDim.x;
    for (; i < (long long)NDTOT; i += stride) {
        const int c = (int)(i & (DDIM - 1));
        const float m   = csum[c] * inv_n;
        const float var = csq[c] * inv_n - m * m;
        float y = (src[i] - m) * rsqrtf(var + 1e-5f) * g[c] + b[c];
        if (relu) y = fmaxf(y, 0.0f);
        dst[i] = y;
    }
}

// ---------------------------------------------------------------------------
extern "C" void kernel_launch(void* const* d_in, const int* in_sizes, int n_in,
                              void* d_out, int out_size, void* d_ws, size_t ws_size,
                              hipStream_t stream) {
    const float* x   = (const float*)d_in[0];
    const int*   ei  = (const int*)d_in[1];
    const float* W1  = (const float*)d_in[2];
    const float* b1  = (const float*)d_in[3];
    const float* g1  = (const float*)d_in[4];
    const float* be1 = (const float*)d_in[5];
    const float* W2  = (const float*)d_in[6];
    const float* b2  = (const float*)d_in[7];
    const float* g2  = (const float*)d_in[8];
    const float* be2 = (const float*)d_in[9];
    const float* ng  = (const float*)d_in[10];
    const float* nb  = (const float*)d_in[11];

    const int* srcI = ei;        // edge_index[0,:]
    const int* dstI = ei + EE;   // edge_index[1,:]

    float* out = (float*)d_out;                    // also h ping buffer

    // Workspace layout (bf16 packs first for 32B alignment of v16bf loads).
    __bf16* Wp1 = (__bf16*)d_ws;                   // 512 KB
    __bf16* Wp2 = Wp1 + (size_t)DDIM * DDIM;       // 512 KB
    float*  csum = (float*)(Wp2 + (size_t)DDIM * DDIM);
    float*  csq  = csum + DDIM;
    float*  agg  = csq + DDIM;                     // [N,D]
    float*  P    = agg + (size_t)NDTOT;            // [N,D]

    const int GEMM_BLOCKS = NN / 16;               // 3125, exact
    const int PACK_BLOCKS = ((DDIM / 32) * DDIM * 2 + 255) / 256;  // 64

    const float* h = x;
    for (int i = 0; i < NL; ++i) {
        const int last = (i == NL - 1);
        const size_t wOff = (size_t)i * DDIM * DDIM;
        const size_t vOff = (size_t)i * DDIM;

        // Pre-pack this layer's weights to bf16 fragment order (L2-resident).
        k_packW<<<PACK_BLOCKS, 256, 0, stream>>>(W1 + wOff, Wp1);
        k_packW<<<PACK_BLOCKS, 256, 0, stream>>>(W2 + wOff, Wp2);

        // agg = segment_sum(h[src], dst)
        k_zero<<<4096, 256, 0, stream>>>(agg, NDTOT);
        k_scatter<<<8192, 256, 0, stream>>>(h, srcI, dstI, agg);

        // P = (h + agg) @ W1 + b1 ; then BN + ReLU (in place)
        k_gemm_bf16<1><<<GEMM_BLOCKS, 256, 0, stream>>>(h, agg, Wp1, b1 + vOff, P);
        k_zero<<<4, 256, 0, stream>>>(csum, 2 * DDIM);
        k_colstats<<<512, 256, 0, stream>>>(P, csum, csq);
        k_bn<<<8192, 256, 0, stream>>>(P, P, csum, csq, g1 + vOff, be1 + vOff, 1);

        // out = P @ W2 + b2 ; BN(g2,be2) (+ReLU unless last)
        k_gemm_bf16<0><<<GEMM_BLOCKS, 256, 0, stream>>>(P, nullptr, Wp2, b2 + vOff, out);
        k_zero<<<4, 256, 0, stream>>>(csum, 2 * DDIM);
        k_colstats<<<512, 256, 0, stream>>>(out, csum, csq);
        k_bn<<<8192, 256, 0, stream>>>(out, out, csum, csq, g2 + vOff, be2 + vOff, last ? 0 : 1);

        // outer BN(ng,nb) (+ReLU unless last) -- in place in d_out
        k_zero<<<4, 256, 0, stream>>>(csum, 2 * DDIM);
        k_colstats<<<512, 256, 0, stream>>>(out, csum, csq);
        k_bn<<<8192, 256, 0, stream>>>(out, out, csum, csq, ng + vOff, nb + vOff, last ? 0 : 1);

        h = out;
    }
}
